// QKVAttentionM_18021682775001
// MI455X (gfx1250) — matrix operand order repaired and checked by reading, not compiled, not run on hardware
//
#include <hip/hip_runtime.h>
#include <hip/hip_bf16.h>

typedef __attribute__((ext_vector_type(16))) _Float16 v16h;
typedef __attribute__((ext_vector_type(8)))  _Float16 v8h;
typedef __attribute__((ext_vector_type(8)))  float    v8f;
typedef __attribute__((ext_vector_type(4)))  unsigned int u32x4;
typedef __attribute__((ext_vector_type(8)))  int          i32x8;
typedef __attribute__((ext_vector_type(4)))  int          i32x4;

#define B_   2
#define S_   2048
#define D_   1024
#define H_   16
#define HD_  64
#define NTOK (B_*S_)   /* 4096 */
#define BH_  (B_*H_)   /* 32 */

// ---------------------------------------------------------------- fp32 -> f16
__global__ void cvt_f32_f16(const float* __restrict__ in,
                            _Float16* __restrict__ out, int n) {
  int i = blockIdx.x * blockDim.x + threadIdx.x;
  int stride = gridDim.x * blockDim.x;
  for (; i < n; i += stride) out[i] = (_Float16)in[i];
}

// ------------------------------------------------------ TDM 1-D tile -> LDS
// Contiguous copy of nelem f16 elements from global to LDS byte offset,
// descriptor per CDNA5 ISA D# (group0: count/lds/global/type, group1: dims).
__device__ inline void tdm_load_1d(unsigned lds_byte_off, const void* gaddr,
                                   unsigned nelem) {
  unsigned long long ga = (unsigned long long)(uintptr_t)gaddr;
  u32x4 g0;
  g0[0] = 1u;                                   // count=1, user descriptor
  g0[1] = lds_byte_off;                         // lds_addr (bytes)
  g0[2] = (unsigned)ga;                         // global_addr[31:0]
  g0[3] = (unsigned)(ga >> 32) | (2u << 30);    // global_addr[56:32] | type=2
  i32x8 g1;
  g1[0] = (int)(1u << 16);                      // wg_mask=0, data_size=1 (2B)
  g1[1] = (int)((nelem & 0xFFFFu) << 16);       // tensor_dim0[15:0]
  g1[2] = (int)(((nelem >> 16) & 0xFFFFu) | (1u << 16)); // dim0 hi | tensor_dim1=1
  g1[3] = (int)((nelem & 0xFFFFu) << 16);       // tile_dim0 = nelem
  g1[4] = 0;                                    // tile_dim1/2 unused
  g1[5] = (int)nelem;                           // tensor_dim0_stride (lo)
  g1[6] = 0;
  g1[7] = 0;
  i32x4 z4 = {0, 0, 0, 0};
#if defined(__clang_major__) && (__clang_major__ >= 23)
  i32x8 z8 = {0, 0, 0, 0, 0, 0, 0, 0};
  __builtin_amdgcn_tensor_load_to_lds(g0, g1, z4, z4, z8, 0);
#else
  __builtin_amdgcn_tensor_load_to_lds(g0, g1, z4, z4, 0);
#endif
}

// -------------------------------------------------- QKV projection (x @ W^T + b)
__global__ __launch_bounds__(256)
void qkv_gemm(const _Float16* __restrict__ X, const _Float16* __restrict__ W,
              const float* __restrict__ bias, _Float16* __restrict__ out) {
  const int lane = threadIdx.x & 31;
  const int wave = threadIdx.x >> 5;
  const int tile = blockIdx.x * 8 + wave;       // 16384 tiles total
  const int tm = tile >> 6;                     // 256 M-tiles (tokens)
  const int tn = tile & 63;                     // 64 N-tiles (features)
  const int l15 = lane & 15;
  const int hi  = lane >> 4;

  const _Float16* ap = X + (size_t)(tm * 16 + l15) * D_ + hi * 16;
  const _Float16* bp = W + (size_t)(tn * 16 + l15) * D_ + hi * 16;

  v8f acc = {};
  for (int k0 = 0; k0 < D_; k0 += 32) {
    v16h a;
    { union { v16h v; v8h h[2]; } u;
      u.h[0] = *(const v8h*)(ap + k0);
      u.h[1] = *(const v8h*)(ap + k0 + 8);
      a = u.v; }
    v16h b = *(const v16h*)(bp + k0);
    acc = __builtin_amdgcn_wmma_f32_16x16x32_f16(false, a, false, b,
                                                 (short)0, acc, false, false);
  }

  const int F  = tn * 16 + l15;                  // feature in [0,1024)
  const float bb = bias[F];
  const int hh = F >> 6, dd = F & 63;
  for (int r = 0; r < 8; ++r) {
    int R = tm * 16 + r + hi * 8;                // token in [0,4096)
    int bat = R >> 11, s = R & (S_ - 1);
    out[(((size_t)(bat * H_ + hh) * S_ + s) << 6) + dd] = (_Float16)(acc[r] + bb);
  }
}

// ----------------------------------------------- qM = gelu(q @ M)  (exact GELU)
__global__ __launch_bounds__(256)
void qm_gemm(const _Float16* __restrict__ Q, const float* __restrict__ M,
             _Float16* __restrict__ out) {
  __shared__ _Float16 Mt[64 * 64];               // Mt[n][k] = M[k][n]
  for (int i = threadIdx.x; i < 64 * 64; i += 256) {
    int k = i >> 6, n = i & 63;
    Mt[n * 64 + k] = (_Float16)M[i];
  }
  __syncthreads();

  const int lane = threadIdx.x & 31;
  const int wave = threadIdx.x >> 5;
  const int tile = blockIdx.x * 8 + wave;        // 16384 tiles
  const int tm = tile >> 2;                      // 4096 M-tiles
  const int tn = tile & 3;                       // 4 N-tiles
  const int l15 = lane & 15, hi = lane >> 4;

  const _Float16* ap = Q + (size_t)(tm * 16 + l15) * 64 + hi * 16;
  const _Float16* bp = Mt + (tn * 16 + l15) * 64 + hi * 16;

  v8f acc = {};
  for (int k0 = 0; k0 < 64; k0 += 32) {
    v16h a;
    { union { v16h v; v8h h[2]; } u;
      u.h[0] = *(const v8h*)(ap + k0);
      u.h[1] = *(const v8h*)(ap + k0 + 8);
      a = u.v; }
    v16h b = *(const v16h*)(bp + k0);
    acc = __builtin_amdgcn_wmma_f32_16x16x32_f16(false, a, false, b,
                                                 (short)0, acc, false, false);
  }
  for (int r = 0; r < 8; ++r) {
    int R = tm * 16 + r + hi * 8;
    float x = acc[r];
    float g = 0.5f * x * (1.0f + erff(x * 0.70710678118f));  // exact GELU
    out[(size_t)R * 64 + tn * 16 + l15] = (_Float16)g;
  }
}

// -------------------------------------------------------- flash attention core
// One block = 128 query rows of one (b,h); each wave owns 16 rows.
// K tiles: TDM tensor_load_to_lds, double-buffered on TENSORcnt.
// V tiles: manual transpose staging (TDM cannot transpose), prefetched.
// LDS layout (halves): Kt0 [0,4096) Kt1 [4096,8192) Vt [8192,12288) Pl [12288,20480)
__global__ __launch_bounds__(256)
void attn(const _Float16* __restrict__ QM, const _Float16* __restrict__ K,
          const _Float16* __restrict__ V, float* __restrict__ out) {
  __shared__ _Float16 smem[20480];               // 40 KB
  _Float16* Vt = smem + 8192;                    // [dim][key]
  _Float16* Pl = smem + 12288;                   // per-wave P scratch

  const int bh   = blockIdx.y;                   // 0..31
  const int qb   = blockIdx.x;                   // 0..15 (128 rows each)
  const int lane = threadIdx.x & 31;
  const int wave = threadIdx.x >> 5;
  const int l15  = lane & 15, hi = lane >> 4;
  const int bat  = bh >> 4, hh = bh & 15;
  const int NK   = S_ / 64;

  // Q A-fragments for this wave's 16 rows (K = 64 dims -> 2 frags), kept live.
  const int qrow = qb * 128 + wave * 16 + l15;
  const _Float16* qp = QM + ((size_t)bh * S_ + qrow) * 64 + hi * 16;
  v16h qa[2];
  for (int ks = 0; ks < 2; ++ks) {
    union { v16h v; v8h h[2]; } u;
    u.h[0] = *(const v8h*)(qp + ks * 32);
    u.h[1] = *(const v8h*)(qp + ks * 32 + 8);
    qa[ks] = u.v;
  }

  float mrow[8], lrow[8];
  v8f o[4] = {};
  for (int r = 0; r < 8; ++r) { mrow[r] = -1e30f; lrow[r] = 0.0f; }

  const _Float16* kbase = K + (size_t)bh * S_ * 64;
  const _Float16* vbase = V + (size_t)bh * S_ * 64;

  // Prologue: DMA first K tile into buffer 0 (wave 0 only; wave-uniform branch).
  if (threadIdx.x < 32) {
    tdm_load_1d(0u, kbase, 64 * 64);
  }

  for (int kb = 0; kb < NK; ++kb) {
    const int cur = kb & 1;

    // ---- stage V tile transposed into LDS (all threads)
    {
      const int t = threadIdx.x;
      const _Float16* gv = vbase + kb * 64 * 64;
      int key = t >> 2, d0 = (t & 3) * 16;
      v8h a0 = *(const v8h*)(gv + key * 64 + d0);
      v8h a1 = *(const v8h*)(gv + key * 64 + d0 + 8);
      for (int j = 0; j < 8; ++j) Vt[(d0 + j) * 64 + key]     = a0[j];
      for (int j = 0; j < 8; ++j) Vt[(d0 + 8 + j) * 64 + key] = a1[j];
      // prefetch next V tile into cache hierarchy
      if (kb + 1 < NK)
        __builtin_prefetch(vbase + (kb + 1) * 64 * 64 + t * 16, 0, 1);
    }

    // ---- TDM: issue next K tile into the other buffer, wait for current
    if (threadIdx.x < 32) {
      if (kb + 1 < NK) {
        tdm_load_1d((unsigned)((cur ^ 1) * 8192), kbase + (kb + 1) * 64 * 64,
                    64 * 64);
        __builtin_amdgcn_s_wait_tensorcnt((short)1);  // current tile done
      } else {
        __builtin_amdgcn_s_wait_tensorcnt((short)0);
      }
    }
    __syncthreads();

    const _Float16* Kt = smem + cur * 4096;        // [key][dim]

    // ---- scores: S = qM (16x64) @ K^T (64x64) -> 4 tiles of 16x16
    v8f sc[4];
    for (int t = 0; t < 4; ++t) {
      v8f c = {};
      const _Float16* kp = Kt + (t * 16 + l15) * 64 + hi * 16;
      for (int ks = 0; ks < 2; ++ks) {
        v16h b = *(const v16h*)(kp + ks * 32);
        c = __builtin_amdgcn_wmma_f32_16x16x32_f16(false, qa[ks], false, b,
                                                   (short)0, c, false, false);
      }
      sc[t] = c;
    }

    // ---- online softmax (row stats reduced across 16-lane half-groups)
    for (int r = 0; r < 8; ++r) {
      float mx = fmaxf(fmaxf(sc[0][r], sc[1][r]), fmaxf(sc[2][r], sc[3][r]));
      for (int m = 1; m < 16; m <<= 1) mx = fmaxf(mx, __shfl_xor(mx, m, 32));
      float mnew  = fmaxf(mrow[r], mx);
      float alpha = __expf(mrow[r] - mnew);
      mrow[r] = mnew;
      float rs = 0.0f;
      for (int t = 0; t < 4; ++t) {
        float p = __expf(sc[t][r] - mnew);
        sc[t][r] = p;
        rs += p;
      }
      for (int m = 1; m < 16; m <<= 1) rs += __shfl_xor(rs, m, 32);
      lrow[r] = lrow[r] * alpha + rs;
      for (int dt = 0; dt < 4; ++dt) o[dt][r] *= alpha;
    }

    // ---- relayout P (C-frag) -> A-frag via per-wave LDS scratch
    _Float16* pw = Pl + wave * 1024;
    for (int t = 0; t < 4; ++t)
      for (int r = 0; r < 8; ++r)
        pw[(r + hi * 8) * 64 + t * 16 + l15] = (_Float16)sc[t][r];
    asm volatile("s_wait_dscnt 0" ::: "memory");  // same-wave DS RAW fence

    // ---- O += P (16x64) @ V (64x64)
    for (int ks = 0; ks < 2; ++ks) {
      v16h pa;
      { union { v16h v; v8h h[2]; } u;
        const _Float16* pp = pw + l15 * 64 + ks * 32 + hi * 16;
        u.h[0] = *(const v8h*)(pp);
        u.h[1] = *(const v8h*)(pp + 8);
        pa = u.v; }
      for (int dt = 0; dt < 4; ++dt) {
        const _Float16* vp = Vt + (dt * 16 + l15) * 64 + ks * 32 + hi * 16;
        v16h b = *(const v16h*)(vp);
        o[dt] = __builtin_amdgcn_wmma_f32_16x16x32_f16(false, pa, false, b,
                                                       (short)0, o[dt], false, false);
      }
    }
    __syncthreads();
  }

  // ---- epilogue: normalize by row sum, write [B,S,H,HD] fp32
  for (int r = 0; r < 8; ++r) {
    int srow = qb * 128 + wave * 16 + r + hi * 8;
    float inv = 1.0f / lrow[r];
    for (int dt = 0; dt < 4; ++dt) {
      int dd = dt * 16 + l15;
      out[(((size_t)bat * S_ + srow) * H_ + hh) * HD_ + dd] = o[dt][r] * inv;
    }
  }
}

// ---------------------------------------------------------------------- launch
extern "C" void kernel_launch(void* const* d_in, const int* in_sizes, int n_in,
                              void* d_out, int out_size, void* d_ws, size_t ws_size,
                              hipStream_t stream) {
  (void)in_sizes; (void)n_in; (void)out_size; (void)ws_size;
  const float* x  = (const float*)d_in[0];
  const float* Wq = (const float*)d_in[1];
  const float* bq = (const float*)d_in[2];
  const float* Wk = (const float*)d_in[3];
  const float* bk = (const float*)d_in[4];
  const float* Wv = (const float*)d_in[5];
  const float* bv = (const float*)d_in[6];
  const float* M  = (const float*)d_in[7];

  // Workspace carve-up (f16 halves): ~46 MB total.
  _Float16* xh  = (_Float16*)d_ws;
  _Float16* wqh = xh  + (size_t)NTOK * D_;
  _Float16* wkh = wqh + (size_t)D_ * D_;
  _Float16* wvh = wkh + (size_t)D_ * D_;
  _Float16* qh  = wvh + (size_t)D_ * D_;       // split-head [BH,S,64]
  _Float16* kh  = qh  + (size_t)NTOK * D_;
  _Float16* vh  = kh  + (size_t)NTOK * D_;
  _Float16* qmh = vh  + (size_t)NTOK * D_;

  cvt_f32_f16<<<512, 256, 0, stream>>>(x,  xh,  NTOK * D_);
  cvt_f32_f16<<<256, 256, 0, stream>>>(Wq, wqh, D_ * D_);
  cvt_f32_f16<<<256, 256, 0, stream>>>(Wk, wkh, D_ * D_);
  cvt_f32_f16<<<256, 256, 0, stream>>>(Wv, wvh, D_ * D_);

  qkv_gemm<<<2048, 256, 0, stream>>>(xh, wqh, bq, qh);
  qkv_gemm<<<2048, 256, 0, stream>>>(xh, wkh, bk, kh);
  qkv_gemm<<<2048, 256, 0, stream>>>(xh, wvh, bv, vh);

  qm_gemm<<<2048, 256, 0, stream>>>(qh, M, qmh);

  attn<<<dim3(16, 32), 256, 0, stream>>>(qmh, kh, vh, (float*)d_out);
}
